// GATLayer_11630771437810
// MI455X (gfx1250) — matrix operand order repaired and checked by reading, not compiled, not run on hardware
//
#include <hip/hip_runtime.h>
#include <hip/hip_bf16.h>

#define N_ROWS 8192
#define IN_F   512
#define OUT_F  256
#define SLOPE  0.1f

typedef __attribute__((ext_vector_type(16))) __bf16          v16bf;
typedef __attribute__((ext_vector_type(8)))  float           v8f;
typedef __attribute__((ext_vector_type(8)))  unsigned int    v8u;

__device__ __forceinline__ unsigned short bf16r(float f) {
    unsigned u = __builtin_bit_cast(unsigned, f);
    return (unsigned short)((u + 0x7FFFu + ((u >> 16) & 1u)) >> 16);
}
__device__ __forceinline__ float lrelu(float x) { return x > 0.f ? x : SLOPE * x; }

// Assemble a v16bf A/B fragment from two 16-byte chunks.
__device__ __forceinline__ v16bf pack_frag(uint4 lo, uint4 hi) {
    v8u au;
    au[0] = lo.x; au[1] = lo.y; au[2] = lo.z; au[3] = lo.w;
    au[4] = hi.x; au[5] = hi.y; au[6] = hi.z; au[7] = hi.w;
    return __builtin_bit_cast(v16bf, au);
}

// ---------------------------------------------------------------------------
// Kernel 0a: W (512x256 f32) -> WbT (256x512 bf16, transposed)
// ---------------------------------------------------------------------------
__global__ void __launch_bounds__(256)
k_convW(const float* __restrict__ W, unsigned short* __restrict__ WbT) {
    int idx = blockIdx.x * 256 + threadIdx.x;   // 131072 total
    int k = idx >> 8;
    int n = idx & 255;
    WbT[n * IN_F + k] = bf16r(W[k * OUT_F + n]);
}

// ---------------------------------------------------------------------------
// Kernel 0b: doc (8192x512 f32) -> docb (bf16, same layout). Streaming convert
// so the GEMM hot loop has no f32->bf16 VALU chains.
// ---------------------------------------------------------------------------
__global__ void __launch_bounds__(256)
k_conv_doc(const float* __restrict__ doc, unsigned short* __restrict__ docb) {
    int idx = (blockIdx.x * 256 + threadIdx.x) * 4;   // 4096 blocks cover 4.19M
    float4 x = *(const float4*)(doc + idx);
    unsigned u0 = (unsigned)bf16r(x.x) | ((unsigned)bf16r(x.y) << 16);
    unsigned u1 = (unsigned)bf16r(x.z) | ((unsigned)bf16r(x.w) << 16);
    *(uint2*)(docb + idx) = make_uint2(u0, u1);
}

// ---------------------------------------------------------------------------
// Kernel 1: h = doc @ W + W_b via v_wmma_f32_16x16x32_bf16.
// Block = 128 threads (4 waves) covering 16 rows x 256 cols; K loop 512/32.
// Emits h (f32) and hbT (bf16, transposed 256x8192).
// ---------------------------------------------------------------------------
__global__ void __launch_bounds__(128)
k_gemm_h(const unsigned short* __restrict__ docb,
         const unsigned short* __restrict__ WbT,
         const float* __restrict__ Wb, float* __restrict__ h,
         unsigned short* __restrict__ hbT) {
    const int tid    = threadIdx.x;
    const int lane   = tid & 31;
    const int wave   = tid >> 5;
    const int mbase  = blockIdx.x * 16;
    const int mL     = lane & 15;
    const int hiHalf = lane >> 4;
    const int klo    = hiHalf * 16;      // A-frag K offset (elements)
    const int kboff  = hiHalf * 16;      // B-frag K offset (elements)
    const int row    = mbase + mL;
    const int ncol0  = wave * 64;

    v8f acc[4];
    for (int t = 0; t < 4; ++t) {
        float b = Wb[ncol0 + t * 16 + mL];
        for (int r = 0; r < 8; ++r) acc[t][r] = b;
    }

    const unsigned short* dp = docb + row * IN_F;
    for (int kb = 0; kb < IN_F; kb += 32) {
        uint4 lo = *(const uint4*)(dp + kb + klo);
        uint4 hi = *(const uint4*)(dp + kb + klo + 8);
        v16bf afrag = pack_frag(lo, hi);
        for (int t = 0; t < 4; ++t) {
            int col = ncol0 + t * 16 + mL;
            v16bf bfrag = *(const v16bf*)(WbT + col * IN_F + kb + kboff);
            acc[t] = __builtin_amdgcn_wmma_f32_16x16x32_bf16(
                false, afrag, false, bfrag, (short)0, acc[t], false, false);
        }
    }

    for (int t = 0; t < 4; ++t) {
        int col = ncol0 + t * 16 + mL;
        for (int r = 0; r < 8; ++r) {
            int rr = mbase + hiHalf * 8 + r;
            float v = acc[t][r];
            h[rr * OUT_F + col]    = v;
            hbT[col * N_ROWS + rr] = bf16r(v);
        }
    }
}

// ---------------------------------------------------------------------------
// Kernel 2: s1 = h@a1, s2 = h@a2. One wave per row.
// ---------------------------------------------------------------------------
__global__ void __launch_bounds__(256)
k_s12(const float* __restrict__ h, const float* __restrict__ a,
      float* __restrict__ s1, float* __restrict__ s2) {
    int lane = threadIdx.x & 31;
    int wave = threadIdx.x >> 5;
    int row  = blockIdx.x * 8 + wave;
    const float* hr = h + row * OUT_F;
    float v1 = 0.f, v2 = 0.f;
    for (int c = lane; c < OUT_F; c += 32) {
        float hv = hr[c];
        v1 += hv * a[c];
        v2 += hv * a[OUT_F + c];
    }
    for (int off = 16; off > 0; off >>= 1) {
        v1 += __shfl_xor(v1, off, 32);
        v2 += __shfl_xor(v2, off, 32);
    }
    if (lane == 0) { s1[row] = v1; s2[row] = v2; }
}

// ---------------------------------------------------------------------------
// Kernel 3: global max over s2 (lrelu monotone => rowmax = lrelu(s1+max_s2+ab))
// ---------------------------------------------------------------------------
__global__ void __launch_bounds__(256)
k_max(const float* __restrict__ s2, float* __restrict__ maxs2) {
    __shared__ float red[256];
    int t = threadIdx.x;
    float m = -3.4e38f;
    for (int i = t; i < N_ROWS; i += 256) m = fmaxf(m, s2[i]);
    red[t] = m; __syncthreads();
    for (int s = 128; s > 0; s >>= 1) {
        if (t < s) red[t] = fmaxf(red[t], red[t + s]);
        __syncthreads();
    }
    if (t == 0) maxs2[0] = red[0];
}

// ---------------------------------------------------------------------------
// Kernel 4: per-row softmax denominator -> rmax[i], isum[i].
// ---------------------------------------------------------------------------
__global__ void __launch_bounds__(256)
k_rowsum(const float* __restrict__ s1, const float* __restrict__ s2,
         const float* __restrict__ maxs2, const float* __restrict__ a_b,
         float* __restrict__ rmax, float* __restrict__ isum) {
    __shared__ float red[256];
    int t = threadIdx.x;
    int row = blockIdx.x;
    float ab  = a_b[0];
    float s1r = s1[row];
    float rm  = lrelu(s1r + maxs2[0] + ab);
    float acc = 0.f;
    for (int j = t; j < N_ROWS; j += 256)
        acc += __expf(lrelu(s1r + s2[j] + ab) - rm);
    red[t] = acc; __syncthreads();
    for (int s = 128; s > 0; s >>= 1) {
        if (t < s) red[t] += red[t + s];
        __syncthreads();
    }
    if (t == 0) { rmax[row] = rm; isum[row] = 1.0f / red[0]; }
}

// ---------------------------------------------------------------------------
// Kernel 5 (fused): per 64-wide K slab, compute 16x64 att tile -> f32 to d_out
// + bf16 to LDS; each wave ds_loads two A-fragments and runs 8 WMMAs
// (2 K-subtiles x 4 N-tiles); out = lrelu(att@h) at the end.
// Block = 128 threads / 4 waves, 16-row strip, 128 slab iterations.
// ---------------------------------------------------------------------------
__global__ void __launch_bounds__(128)
k_att_out(const unsigned short* __restrict__ hbT,
          const float* __restrict__ s1, const float* __restrict__ s2,
          const float* __restrict__ rmax, const float* __restrict__ isum,
          const float* __restrict__ a_b,
          float* __restrict__ out, float* __restrict__ att) {
    const int ST = 72;   // padded ushort stride per att row (144 B, 16B-aligned)
    __shared__ float s1s[16], rms[16], iss[16];
    __shared__ __align__(16) unsigned short att_s[16 * ST];

    const int tid    = threadIdx.x;
    const int lane   = tid & 31;
    const int wave   = tid >> 5;
    const int mbase  = blockIdx.x * 16;
    const int mL     = lane & 15;
    const int hiHalf = lane >> 4;
    const int klo    = hiHalf * 16;
    const int kboff  = hiHalf * 16;
    const int ncol0  = wave * 64;
    const float ab   = a_b[0];

    if (tid < 16) {
        s1s[tid] = s1[mbase + tid];
        rms[tid] = rmax[mbase + tid];
        iss[tid] = isum[mbase + tid];
    }
    __syncthreads();

    // att-tile compute assignment: thread -> (row cm, 8 cols from ck)
    const int cm = tid >> 3;          // 0..15
    const int ck = (tid & 7) * 8;     // 0,8,...,56
    const float cs1 = s1s[cm], crm = rms[cm], cis = iss[cm];
    float* attRow = att + (size_t)(mbase + cm) * N_ROWS;

    v8f acc[4] = {};

    for (int jb = 0; jb < N_ROWS; jb += 64) {
        const float* s2p = s2 + jb + ck;
        float4 sa = *(const float4*)(s2p);
        float4 sb = *(const float4*)(s2p + 4);
        float e0 = __expf(lrelu(cs1 + sa.x + ab) - crm) * cis;
        float e1 = __expf(lrelu(cs1 + sa.y + ab) - crm) * cis;
        float e2 = __expf(lrelu(cs1 + sa.z + ab) - crm) * cis;
        float e3 = __expf(lrelu(cs1 + sa.w + ab) - crm) * cis;
        float e4 = __expf(lrelu(cs1 + sb.x + ab) - crm) * cis;
        float e5 = __expf(lrelu(cs1 + sb.y + ab) - crm) * cis;
        float e6 = __expf(lrelu(cs1 + sb.z + ab) - crm) * cis;
        float e7 = __expf(lrelu(cs1 + sb.w + ab) - crm) * cis;
        float* ao = attRow + jb + ck;
        *(float4*)(ao)     = make_float4(e0, e1, e2, e3);
        *(float4*)(ao + 4) = make_float4(e4, e5, e6, e7);
        unsigned short* ap = att_s + cm * ST + ck;
        ap[0] = bf16r(e0); ap[1] = bf16r(e1); ap[2] = bf16r(e2); ap[3] = bf16r(e3);
        ap[4] = bf16r(e4); ap[5] = bf16r(e5); ap[6] = bf16r(e6); ap[7] = bf16r(e7);
        __syncthreads();

        // Two A fragments (K sub-slabs 0..31 and 32..63) for row mL.
        const unsigned short* ar = att_s + mL * ST;
        v16bf afrag0 = pack_frag(*(const uint4*)(ar + klo),
                                 *(const uint4*)(ar + klo + 8));
        v16bf afrag1 = pack_frag(*(const uint4*)(ar + 32 + klo),
                                 *(const uint4*)(ar + 32 + klo + 8));

        for (int t = 0; t < 4; ++t) {
            int col = ncol0 + t * 16 + mL;
            const unsigned short* hp = hbT + (size_t)col * N_ROWS + jb + kboff;
            v16bf b0 = *(const v16bf*)(hp);
            v16bf b1 = *(const v16bf*)(hp + 32);
            acc[t] = __builtin_amdgcn_wmma_f32_16x16x32_bf16(
                false, afrag0, false, b0, (short)0, acc[t], false, false);
            acc[t] = __builtin_amdgcn_wmma_f32_16x16x32_bf16(
                false, afrag1, false, b1, (short)0, acc[t], false, false);
        }
        __syncthreads();
    }

    for (int t = 0; t < 4; ++t) {
        int col = ncol0 + t * 16 + mL;
        for (int r = 0; r < 8; ++r) {
            int rr = mbase + hiHalf * 8 + r;
            out[rr * OUT_F + col] = lrelu(acc[t][r]);
        }
    }
}

// ---------------------------------------------------------------------------
extern "C" void kernel_launch(void* const* d_in, const int* in_sizes, int n_in,
                              void* d_out, int out_size, void* d_ws, size_t ws_size,
                              hipStream_t stream) {
    const float* doc = (const float*)d_in[0];
    const float* W   = (const float*)d_in[1];
    const float* Wb  = (const float*)d_in[2];
    const float* a   = (const float*)d_in[3];
    const float* ab  = (const float*)d_in[4];

    char* ws = (char*)d_ws;
    unsigned short* WbT  = (unsigned short*)(ws);                     // 256 KB
    unsigned short* hbT  = (unsigned short*)(ws + 262144);            // 4 MB
    float*          h    = (float*)(ws + 4456448);                    // 8 MB
    unsigned short* docb = (unsigned short*)(ws + 12845056);          // 8 MB
    float* s1    = (float*)(ws + 21233664);
    float* s2    = (float*)(ws + 21266432);
    float* rmaxv = (float*)(ws + 21299200);
    float* isumv = (float*)(ws + 21331968);
    float* maxs2 = (float*)(ws + 21364736);

    float* out = (float*)d_out;
    float* att = out + (size_t)N_ROWS * OUT_F;

    k_convW   <<<512, 256, 0, stream>>>(W, WbT);
    k_conv_doc<<<4096, 256, 0, stream>>>(doc, docb);
    k_gemm_h  <<<512, 128, 0, stream>>>(docb, WbT, Wb, h, hbT);
    k_s12     <<<1024, 256, 0, stream>>>(h, a, s1, s2);
    k_max     <<<1, 256, 0, stream>>>(s2, maxs2);
    k_rowsum  <<<N_ROWS, 256, 0, stream>>>(s1, s2, maxs2, ab, rmaxv, isumv);
    k_att_out <<<512, 128, 0, stream>>>(hbT, s1, s2, rmaxv, isumv, ab, out, att);
}
